// DeepseekV32Indexer_48326972015356
// MI455X (gfx1250) — matrix operand order repaired and checked by reading, not compiled, not run on hardware
//
#include <hip/hip_runtime.h>
#include <stdint.h>

// ---------------------------------------------------------------------------
// Types for CDNA5 WMMA (wave32): V_WMMA_F32_16X16X32_BF16
// ---------------------------------------------------------------------------
typedef __attribute__((ext_vector_type(16))) __bf16 v16bf;
typedef __attribute__((ext_vector_type(8)))  float  v8f;

struct __attribute__((aligned(16))) U4 { unsigned int a, b, c, d; };
union FragU { v16bf bf; U4 q[2]; };

__device__ __forceinline__ unsigned short f2bf(float f) {
    __bf16 h = (__bf16)f;                 // native v_cvt f32->bf16 (RNE)
    return __builtin_bit_cast(unsigned short, h);
}

// CDNA5 async global->LDS copy (ASYNCcnt-tracked, no VGPR data round trip).
// lds_off = low 32 bits of generic shared pointer (HW adds wave LDS_BASE).
__device__ __forceinline__ void async_copy_b128(unsigned int lds_off, const void* g) {
    asm volatile("global_load_async_to_lds_b128 %0, %1, off"
                 :: "v"(lds_off), "v"((unsigned long long)(uintptr_t)g)
                 : "memory");
}
__device__ __forceinline__ void async_wait0() {
    asm volatile("s_wait_asynccnt 0x0" ::: "memory");
}

// A fragment (16x32 bf16, MxK). ISA 7.12.2: lane<16 -> row=lane, K {0..7,16..23};
// lane>=16 -> row=lane-16, K {8..15,24..31}. base = tile origin, row-major [row][k].
__device__ __forceinline__ v16bf load_a_frag(const unsigned short* base, int ldk, int lane) {
    const int r = lane & 15, half = lane >> 4;
    const unsigned short* p = base + r * ldk + half * 8;
    FragU f;
    f.q[0] = *(const U4*)(p);        // K +0..7   (or +8..15)
    f.q[1] = *(const U4*)(p + 16);   // K +16..23 (or +24..31)
    return f.bf;
}

// B fragment (32x16 bf16, KxN) loaded from k-major storage [n][k] (B^T row-major).
// lane<16 -> col=lane, K {0..7,16..23}; lane>=16 -> K {8..15,24..31}; 2 consecutive K per VGPR.
__device__ __forceinline__ v16bf load_b_frag(const unsigned short* base, int ldk, int lane) {
    const int r = lane & 15, half = lane >> 4;
    const unsigned short* p = base + r * ldk + half * 8;
    FragU f;
    f.q[0] = *(const U4*)(p);
    f.q[1] = *(const U4*)(p + 16);
    return f.bf;
}

__device__ __forceinline__ v8f wmma_bf16(v16bf a, v16bf b, v8f c) {
    return __builtin_amdgcn_wmma_f32_16x16x32_bf16(false, a, false, b, (short)0, c, false, false);
}

// ---------------------------------------------------------------------------
// Problem constants
// ---------------------------------------------------------------------------
#define S_LEN 4096
#define HID   7168
#define QLR   1536
#define NH    64
#define DD    128
#define RR    64
#define NQ    (NH * DD)          // 8192
#define NCOMB 192                // 128 (wk) + 64 (w_proj)
#define TOPK_N 2048

// ---------------------------------------------------------------------------
// Prep: wq_b [1536][8192] f32 -> wqbT [8192][1536] bf16 (transposed)
// ---------------------------------------------------------------------------
__global__ void prep_wqbT(const float* __restrict__ wqb, unsigned short* __restrict__ wqbT) {
    size_t total = (size_t)QLR * NQ;
    for (size_t i = (size_t)blockIdx.x * blockDim.x + threadIdx.x; i < total;
         i += (size_t)gridDim.x * blockDim.x) {
        size_t k = i >> 13;          // / 8192
        size_t n = i & (NQ - 1);
        wqbT[n * QLR + k] = f2bf(wqb[i]);
    }
}

// Prep: wcombT [192][7168] bf16 ; rows 0..127 = wk columns, rows 128..191 = w_proj columns
__global__ void prep_wcombT(const float* __restrict__ wk, const float* __restrict__ wproj,
                            unsigned short* __restrict__ wcombT) {
    size_t total = (size_t)NCOMB * HID;
    for (size_t i = (size_t)blockIdx.x * blockDim.x + threadIdx.x; i < total;
         i += (size_t)gridDim.x * blockDim.x) {
        int n = (int)(i / HID);
        int k = (int)(i % HID);
        float v = (n < DD) ? wk[(size_t)k * DD + n] : wproj[(size_t)k * NH + (n - DD)];
        wcombT[i] = f2bf(v);
    }
}

// ---------------------------------------------------------------------------
// Kernel A: hs(4096x7168) @ [wk|w_proj](7168x192), M=32 per block, 4 waves.
// B tiles staged via async global->LDS; A tiles converted f32->bf16 in flight.
// Epilogue: LayerNorm+RoPE on k -> bf16 ; weights * H^-0.5 -> f32.
// ---------------------------------------------------------------------------
__launch_bounds__(128)
__global__ void kernelA(const float* __restrict__ hs, const float* __restrict__ cosb,
                        const float* __restrict__ sinb, const float* __restrict__ kgam,
                        const float* __restrict__ kbet, const unsigned short* __restrict__ wcombT,
                        unsigned short* __restrict__ k_out, float* __restrict__ w_out) {
    __shared__ unsigned short Alds[32][32];
    __shared__ unsigned short Blds[NCOMB][32];
    __shared__ float outb[32][NCOMB];
    __shared__ float mu_s[32], rs_s[32];

    const int tid = threadIdx.x, lane = tid & 31, wave = tid >> 5;
    const int s0 = blockIdx.x * 32;

    v8f acc[2][3] = {};

    for (int k0 = 0; k0 < HID; k0 += 32) {
        { // issue async B^T staging: 192 rows x 64B, thread t -> row t (+ row t+128)
            unsigned int l0 = (unsigned int)(uintptr_t)&Blds[tid][0];
            const unsigned char* g0 = (const unsigned char*)(wcombT + (size_t)tid * HID + k0);
            async_copy_b128(l0,      g0);
            async_copy_b128(l0 + 16, g0 + 16);
            async_copy_b128(l0 + 32, g0 + 32);
            async_copy_b128(l0 + 48, g0 + 48);
            if (tid < 64) {
                int rr = tid + 128;
                unsigned int l1 = (unsigned int)(uintptr_t)&Blds[rr][0];
                const unsigned char* g1 = (const unsigned char*)(wcombT + (size_t)rr * HID + k0);
                async_copy_b128(l1,      g1);
                async_copy_b128(l1 + 16, g1 + 16);
                async_copy_b128(l1 + 32, g1 + 32);
                async_copy_b128(l1 + 48, g1 + 48);
            }
        }
        { // stage A: 32x32 f32 -> bf16 (overlaps with async B copy)
            int row = tid >> 2, c0 = (tid & 3) * 8;
            const float* src = hs + (size_t)(s0 + row) * HID + k0 + c0;
            unsigned short* dst = &Alds[row][c0];
#pragma unroll
            for (int i = 0; i < 8; ++i) dst[i] = f2bf(src[i]);
        }
        async_wait0();
        __syncthreads();
        v16bf a0 = load_a_frag(&Alds[0][0], 32, lane);
        v16bf a1 = load_a_frag(&Alds[16][0], 32, lane);
#pragma unroll
        for (int j = 0; j < 3; ++j) {
            v16bf b = load_b_frag(&Blds[wave * 48 + j * 16][0], 32, lane);
            acc[0][j] = wmma_bf16(a0, b, acc[0][j]);
            acc[1][j] = wmma_bf16(a1, b, acc[1][j]);
        }
        __syncthreads();
    }

    { // scatter C (lane=N, vgpr=M per ISA layout)
        const int r = lane & 15, half = lane >> 4;
#pragma unroll
        for (int mt = 0; mt < 2; ++mt)
#pragma unroll
            for (int j = 0; j < 3; ++j) {
                int n = wave * 48 + j * 16 + r;
                int mb = mt * 16 + half * 8;
#pragma unroll
                for (int i = 0; i < 8; ++i) outb[mb + i][n] = acc[mt][j][i];
            }
    }
    __syncthreads();

    if (tid < 32) { // LN stats over first 128 cols
        float m = 0.f;
        for (int d = 0; d < DD; ++d) m += outb[tid][d];
        m *= (1.0f / DD);
        float v = 0.f;
        for (int d = 0; d < DD; ++d) { float x = outb[tid][d] - m; v += x * x; }
        v *= (1.0f / DD);
        mu_s[tid] = m;
        rs_s[tid] = rsqrtf(v + 1e-6f);
    }
    __syncthreads();

    for (int e = tid; e < 32 * DD; e += 128) { // k: LN + RoPE -> bf16
        int row = e >> 7, d = e & (DD - 1);
        int s = s0 + row;
        float m = mu_s[row], rs = rs_s[row];
        float x = (outb[row][d] - m) * rs * kgam[d] + kbet[d];
        float o;
        if (d < RR) {
            int dp = (d < 32) ? d + 32 : d - 32;
            float xp = (outb[row][dp] - m) * rs * kgam[dp] + kbet[dp];
            float rot = (d < 32) ? -xp : xp;
            o = x * cosb[s * RR + d] + rot * sinb[s * RR + d];
        } else o = x;
        k_out[(size_t)s * DD + d] = f2bf(o);
    }
    for (int e = tid; e < 32 * NH; e += 128) { // weights * H^-0.5
        int row = e >> 6, h = e & (NH - 1);
        w_out[(size_t)(s0 + row) * NH + h] = outb[row][DD + h] * 0.125f;
    }
}

// ---------------------------------------------------------------------------
// Kernel B: qw[s,d] = RoPE( sum_h w[s,h] * (q_resid @ wq_b)[s,h,d] )
// M=32 rows/block, q_resid tile resident in LDS (bf16), loop 64 heads x K=1536.
// wq_b tiles staged with async global->LDS copies.
// Dynamic LDS: qr 96KB | wq 8KB | w 8KB | qw 16KB = 128KB.
// ---------------------------------------------------------------------------
extern __shared__ unsigned char smemB[];
__launch_bounds__(128)
__global__ void kernelB(const float* __restrict__ qr, const unsigned short* __restrict__ wqbT,
                        const float* __restrict__ w_in, const float* __restrict__ cosb,
                        const float* __restrict__ sinb, unsigned short* __restrict__ qw_out) {
    unsigned short* qrl = (unsigned short*)smemB;                     // [32][1536]
    unsigned short* wql = (unsigned short*)(smemB + 98304);           // [128][32]
    float* wl = (float*)(smemB + 98304 + 8192);                       // [32][64]
    float* qw = (float*)(smemB + 98304 + 8192 + 8192);                // [32][128]

    const int tid = threadIdx.x, lane = tid & 31, wave = tid >> 5;
    const int s0 = blockIdx.x * 32;

    { // stage resident A tile: 32x1536 f32 -> bf16
        int row = tid >> 2, c0 = (tid & 3) * 384;
        const float* src = qr + (size_t)(s0 + row) * QLR + c0;
        unsigned short* dst = qrl + row * QLR + c0;
        for (int i = 0; i < 384; ++i) dst[i] = f2bf(src[i]);
    }
    for (int e = tid; e < 32 * NH; e += 128) wl[e] = w_in[(size_t)s0 * NH + e];
    for (int e = tid; e < 32 * DD; e += 128) qw[e] = 0.f;
    __syncthreads();

    for (int h = 0; h < NH; ++h) {
        v8f acc[2][2] = {};
        for (int k0 = 0; k0 < QLR; k0 += 32) {
            { // async stage B^T tile: row n=h*128+tid, 64B each
                unsigned int l0 = (unsigned int)(uintptr_t)(wql + tid * 32);
                const unsigned char* g0 =
                    (const unsigned char*)(wqbT + ((size_t)h * DD + tid) * QLR + k0);
                async_copy_b128(l0,      g0);
                async_copy_b128(l0 + 16, g0 + 16);
                async_copy_b128(l0 + 32, g0 + 32);
                async_copy_b128(l0 + 48, g0 + 48);
            }
            async_wait0();
            __syncthreads();
            v16bf a0 = load_a_frag(qrl + k0, QLR, lane);
            v16bf a1 = load_a_frag(qrl + 16 * QLR + k0, QLR, lane);
#pragma unroll
            for (int j = 0; j < 2; ++j) {
                v16bf b = load_b_frag(wql + (wave * 2 + j) * 16 * 32, 32, lane);
                acc[0][j] = wmma_bf16(a0, b, acc[0][j]);
                acc[1][j] = wmma_bf16(a1, b, acc[1][j]);
            }
            __syncthreads();
        }
        { // weighted head accumulation (each (m,n) owned by exactly one lane)
            const int r = lane & 15, half = lane >> 4;
#pragma unroll
            for (int mt = 0; mt < 2; ++mt)
#pragma unroll
                for (int j = 0; j < 2; ++j) {
                    int n = (wave * 2 + j) * 16 + r;
                    int mb = mt * 16 + half * 8;
#pragma unroll
                    for (int i = 0; i < 8; ++i) {
                        int m = mb + i;
                        qw[m * DD + n] += wl[m * NH + h] * acc[mt][j][i];
                    }
                }
        }
    }
    __syncthreads();

    for (int e = tid; e < 32 * DD; e += 128) { // RoPE (linear => valid post-sum)
        int row = e >> 7, d = e & (DD - 1);
        int s = s0 + row;
        float x = qw[row * DD + d];
        float o;
        if (d < RR) {
            int dp = (d < 32) ? d + 32 : d - 32;
            float xp = qw[row * DD + dp];
            float rot = (d < 32) ? -xp : xp;
            o = x * cosb[s * RR + d] + rot * sinb[s * RR + d];
        } else o = x;
        qw_out[(size_t)s * DD + d] = f2bf(o);
    }
}

// ---------------------------------------------------------------------------
// Kernel C: scores = qw(4096x128) @ k^T * D^-0.5 + mask. Fragments straight
// from L2-resident bf16 buffers (both row-major in d == K). M=32, N=128/block.
// ---------------------------------------------------------------------------
__launch_bounds__(128)
__global__ void kernelC(const unsigned short* __restrict__ qwb, const unsigned short* __restrict__ kb,
                        const float* __restrict__ mask, float* __restrict__ scores) {
    const int tid = threadIdx.x, lane = tid & 31, wave = tid >> 5;
    const int s0 = blockIdx.x * 32;
    const int t0 = blockIdx.y * 128;
    v8f acc[2][2] = {};
#pragma unroll
    for (int k0 = 0; k0 < DD; k0 += 32) {
        v16bf a0 = load_a_frag(qwb + (size_t)s0 * DD + k0, DD, lane);
        v16bf a1 = load_a_frag(qwb + (size_t)(s0 + 16) * DD + k0, DD, lane);
#pragma unroll
        for (int j = 0; j < 2; ++j) {
            int n0 = t0 + (wave * 2 + j) * 16;
            v16bf b = load_b_frag(kb + (size_t)n0 * DD + k0, DD, lane);
            acc[0][j] = wmma_bf16(a0, b, acc[0][j]);
            acc[1][j] = wmma_bf16(a1, b, acc[1][j]);
        }
    }
    const float sc = 0.08838834764831845f;   // 128^-0.5
    const int r = lane & 15, half = lane >> 4;
#pragma unroll
    for (int mt = 0; mt < 2; ++mt)
#pragma unroll
        for (int j = 0; j < 2; ++j) {
            int t = t0 + (wave * 2 + j) * 16 + r;
            int mb = s0 + mt * 16 + half * 8;
#pragma unroll
            for (int i = 0; i < 8; ++i) {
                size_t idx = (size_t)(mb + i) * S_LEN + t;
                scores[idx] = acc[mt][j][i] * sc + mask[idx];
            }
        }
}

// ---------------------------------------------------------------------------
// Kernel D: per-row top-2048 via full bitonic sort of 4096 (score,idx) pairs.
// Descending by score, ties broken by ascending index (matches jax top_k).
// ---------------------------------------------------------------------------
__launch_bounds__(256)
__global__ void kernelD(const float* __restrict__ scores, int* __restrict__ out) {
    __shared__ float v[S_LEN];
    __shared__ int ix[S_LEN];
    const int tid = threadIdx.x;
    const int s = blockIdx.x;
    const float* row = scores + (size_t)s * S_LEN;
    for (int i = tid; i < S_LEN; i += 256) { v[i] = row[i]; ix[i] = i; }
    __syncthreads();
    for (int k = 2; k <= S_LEN; k <<= 1) {
        for (int j = k >> 1; j > 0; j >>= 1) {
            for (int t = tid; t < S_LEN / 2; t += 256) {
                int i = ((t & ~(j - 1)) << 1) | (t & (j - 1));
                int p = i | j;
                bool up = ((i & k) == 0);                 // "up" block = descending
                float vi = v[i], vp = v[p];
                int ii = ix[i], ip = ix[p];
                bool pBeforeI = (vp > vi) || (vp == vi && ip < ii);
                if (up == pBeforeI) { v[i] = vp; v[p] = vi; ix[i] = ip; ix[p] = ii; }
            }
            __syncthreads();
        }
    }
    for (int i = tid; i < TOPK_N; i += 256) out[(size_t)s * TOPK_N + i] = ix[i];
}

// ---------------------------------------------------------------------------
// Launch
// ---------------------------------------------------------------------------
extern "C" void kernel_launch(void* const* d_in, const int* in_sizes, int n_in,
                              void* d_out, int out_size, void* d_ws, size_t ws_size,
                              hipStream_t stream) {
    const float* hs   = (const float*)d_in[0];
    const float* qr   = (const float*)d_in[1];
    const float* cosb = (const float*)d_in[2];
    const float* sinb = (const float*)d_in[3];
    const float* mask = (const float*)d_in[4];
    const float* wqb  = (const float*)d_in[5];
    const float* wk   = (const float*)d_in[6];
    const float* kg   = (const float*)d_in[7];
    const float* kb   = (const float*)d_in[8];
    const float* wp   = (const float*)d_in[9];
    int* out = (int*)d_out;

    unsigned char* ws = (unsigned char*)d_ws;
    unsigned short* wqbT   = (unsigned short*)(ws);              // 25,165,824 B
    unsigned short* wcombT = (unsigned short*)(ws + 25165824);   //  2,752,512 B
    unsigned short* kbf    = (unsigned short*)(ws + 27918336);   //  1,048,576 B
    unsigned short* qwbf   = (unsigned short*)(ws + 28966912);   //  1,048,576 B
    float*          wts    = (float*)(ws + 30015488);            //  1,048,576 B
    float*          scores = (float*)(ws + 31064064);            // 67,108,864 B

    prep_wqbT<<<dim3(2048), dim3(256), 0, stream>>>(wqb, wqbT);
    prep_wcombT<<<dim3(512), dim3(256), 0, stream>>>(wk, wp, wcombT);
    kernelA<<<dim3(S_LEN / 32), dim3(128), 0, stream>>>(hs, cosb, sinb, kg, kb, wcombT, kbf, wts);
    hipFuncSetAttribute((const void*)kernelB, hipFuncAttributeMaxDynamicSharedMemorySize, 131072);
    kernelB<<<dim3(S_LEN / 32), dim3(128), 131072, stream>>>(qr, wqbT, wts, cosb, sinb, qwbf);
    kernelC<<<dim3(S_LEN / 32, S_LEN / 128), dim3(128), 0, stream>>>(qwbf, kbf, mask, scores);
    kernelD<<<dim3(S_LEN), dim3(256), 0, stream>>>(scores, out);
    (void)in_sizes; (void)n_in; (void)out_size; (void)ws_size;
}
